// DeepSeekV3Attention_45947560133087
// MI455X (gfx1250) — compile-verified, matrix operand order checked by reading, not run on hardware
//
#include <hip/hip_runtime.h>
#include <hip/hip_bf16.h>

typedef __bf16 bf16;
typedef __attribute__((ext_vector_type(8)))  __bf16 v8bf;
typedef __attribute__((ext_vector_type(16))) __bf16 v16bf;
typedef __attribute__((ext_vector_type(8)))  float  v8f;

#define S_LEN   2048
#define H_DIM   4096
#define N_HEADS 32
#define DR_     64
#define DN_     128
#define DV_     128
#define QD_     192      // DN + DR
#define QR_     1536
#define KR_     512
#define SCALE_F 0.07216878364870323f   // (192)^-0.5

#define WMMA_BF16(a, b, c) \
  __builtin_amdgcn_wmma_f32_16x16x32_bf16(false, (a), false, (b), (short)0, (c), false, false)

// ---------------------------------------------------------------------------
// Async global -> LDS staging (CDNA5, tracked by ASYNCcnt).
// Inline asm form: VDST = per-lane LDS byte address, VADDR = 64-bit global
// address (ISA 15.18.3, GLOBAL_LOAD_ASYNC_TO_LDS_B128).
// ---------------------------------------------------------------------------
__device__ __forceinline__ void async_load16(const bf16* g, bf16* l) {
  unsigned lds = (unsigned)(size_t)(__attribute__((address_space(3))) bf16*)l;
  asm volatile("global_load_async_to_lds_b128 %0, %1, off"
               :: "v"(lds), "v"(g)
               : "memory");
}

__device__ __forceinline__ void wait_async0() {
#if __has_builtin(__builtin_amdgcn_s_wait_asynccnt)
  __builtin_amdgcn_s_wait_asynccnt(0);
#else
  asm volatile("s_wait_asynccnt 0x0" ::: "memory");
#endif
}

// ---------------------------------------------------------------------------
// Fragment loaders (wave32, 16-bit A/B layout per CDNA5 ISA 7.12.2):
//   lanes 0-15:  row = row0+lane,    K = k0+0..7  and k0+16..23
//   lanes 16-31: row = row0+lane-16, K = k0+8..15 and k0+24..31
// ---------------------------------------------------------------------------
__device__ __forceinline__ v16bf frag_b16(const bf16* base, int stride, int row0,
                                          int k0, int lane) {
  int r  = row0 + (lane & 15);
  int ks = k0 + ((lane >> 4) << 3);
  const bf16* p = base + (size_t)r * stride + ks;
  v8bf lo = *(const v8bf*)p;
  v8bf hi = *(const v8bf*)(p + 16);
  return __builtin_shufflevector(lo, hi, 0, 1, 2, 3, 4, 5, 6, 7,
                                 8, 9, 10, 11, 12, 13, 14, 15);
}

__device__ __forceinline__ v16bf frag_f32(const float* base, int stride, int row0,
                                          int k0, int lane) {
  int r  = row0 + (lane & 15);
  int ks = k0 + ((lane >> 4) << 3);
  const float* p = base + (size_t)r * stride + ks;
  v16bf out;
#pragma unroll
  for (int i = 0; i < 8; i++) out[i] = (bf16)p[i];
#pragma unroll
  for (int i = 0; i < 8; i++) out[i + 8] = (bf16)p[i + 16];
  return out;
}

// ---------------------------------------------------------------------------
// bf16 WMMA GEMM: C[M,N] = A[M,K] * B[N,K]^T, f32 accumulate.
// 128x128 block tile, BK=32, double-buffered LDS filled by async loads.
// 8 waves: wave w -> rows (w&3)*32 (2 m-tiles), cols (w>>2)*64 (4 n-tiles).
// Out-of-range A/B rows are simply not fetched: stale LDS only feeds WMMA
// outputs whose stores are bounds-guarded.
// ---------------------------------------------------------------------------
__global__ __launch_bounds__(256) void gemm_bf16_nt(const bf16* __restrict__ A,
                                                    const bf16* __restrict__ B,
                                                    float* __restrict__ C,
                                                    int M, int N, int K) {
  __shared__ bf16 As[2][128][32];
  __shared__ bf16 Bs[2][128][32];
  const int tid  = threadIdx.x;
  const int wid  = tid >> 5;
  const int lane = tid & 31;
  const int bm = blockIdx.y * 128;
  const int bn = blockIdx.x * 128;
  const int wm = (wid & 3) * 32;
  const int wn = (wid >> 2) * 64;

  v8f acc[2][4];
#pragma unroll
  for (int mi = 0; mi < 2; mi++)
#pragma unroll
    for (int ni = 0; ni < 4; ni++)
#pragma unroll
      for (int r = 0; r < 8; r++) acc[mi][ni][r] = 0.0f;

  const int lrow = tid >> 1;
  const int lseg = (tid & 1) * 16;
  const int  grA = bm + lrow;
  const int  grB = bn + lrow;
  const bool okA = grA < M;
  const bool okB = grB < N;
  const bf16* gA = A + (size_t)grA * K + lseg;
  const bf16* gB = B + (size_t)grB * K + lseg;

  // stage first tile
  if (okA) {
    async_load16(gA,     &As[0][lrow][lseg]);
    async_load16(gA + 8, &As[0][lrow][lseg + 8]);
  }
  if (okB) {
    async_load16(gB,     &Bs[0][lrow][lseg]);
    async_load16(gB + 8, &Bs[0][lrow][lseg + 8]);
  }

  const int T = K >> 5;
  for (int t = 0; t < T; t++) {
    wait_async0();        // own tile-t loads landed in LDS
    __syncthreads();      // everyone's loads landed; prev compute done
    if (t + 1 < T) {      // prefetch tile t+1 into the other buffer
      int k0 = (t + 1) << 5;
      int b  = (t + 1) & 1;
      if (okA) {
        async_load16(gA + k0,     &As[b][lrow][lseg]);
        async_load16(gA + k0 + 8, &As[b][lrow][lseg + 8]);
      }
      if (okB) {
        async_load16(gB + k0,     &Bs[b][lrow][lseg]);
        async_load16(gB + k0 + 8, &Bs[b][lrow][lseg + 8]);
      }
    }
    const int b = t & 1;
    v16bf afr[2], bfr[4];
#pragma unroll
    for (int mi = 0; mi < 2; mi++) afr[mi] = frag_b16(&As[b][0][0], 32, wm + 16 * mi, 0, lane);
#pragma unroll
    for (int ni = 0; ni < 4; ni++) bfr[ni] = frag_b16(&Bs[b][0][0], 32, wn + 16 * ni, 0, lane);
#pragma unroll
    for (int mi = 0; mi < 2; mi++)
#pragma unroll
      for (int ni = 0; ni < 4; ni++)
        acc[mi][ni] = WMMA_BF16(afr[mi], bfr[ni], acc[mi][ni]);
  }

#pragma unroll
  for (int mi = 0; mi < 2; mi++)
#pragma unroll
    for (int ni = 0; ni < 4; ni++)
#pragma unroll
      for (int r = 0; r < 8; r++) {
        int row = bm + wm + 16 * mi + r + ((lane >> 4) << 3);
        int col = bn + wn + 16 * ni + (lane & 15);
        if (row < M && col < N) C[(size_t)row * N + col] = acc[mi][ni][r];
      }
}

// ---------------------------------------------------------------------------
// Elementwise / norm / rope kernels
// ---------------------------------------------------------------------------
__global__ void cvt_f32_bf16_kernel(const float* __restrict__ in,
                                    bf16* __restrict__ out, int n) {
  int i = blockIdx.x * blockDim.x + threadIdx.x;
  if (i < n) out[i] = (bf16)in[i];
}

__global__ void rmsnorm_kernel(const float* __restrict__ in,
                               const float* __restrict__ w,
                               bf16* __restrict__ out, int D) {
  __shared__ float red[256];
  const int row = blockIdx.x, tid = threadIdx.x;
  const float* x = in + (size_t)row * D;
  float s = 0.0f;
  for (int i = tid; i < D; i += 256) { float v = x[i]; s += v * v; }
  red[tid] = s;
  __syncthreads();
  for (int st = 128; st > 0; st >>= 1) {
    if (tid < st) red[tid] += red[tid + st];
    __syncthreads();
  }
  float inv = rsqrtf(red[0] / (float)D + 1e-6f);
  for (int i = tid; i < D; i += 256)
    out[(size_t)row * D + i] = (bf16)(x[i] * inv * w[i]);
}

__global__ void kva_process_kernel(const float* __restrict__ kva,
                                   const float* __restrict__ ln_w,
                                   const int* __restrict__ pos,
                                   bf16* __restrict__ ckv,
                                   float* __restrict__ kpe) {
  __shared__ float red[256];
  const int row = blockIdx.x, tid = threadIdx.x;
  const float* xr = kva + (size_t)row * (KR_ + DR_);
  float s = 0.0f;
  for (int i = tid; i < KR_; i += 256) { float v = xr[i]; s += v * v; }
  red[tid] = s;
  __syncthreads();
  for (int st = 128; st > 0; st >>= 1) {
    if (tid < st) red[tid] += red[tid + st];
    __syncthreads();
  }
  float inv = rsqrtf(red[0] / (float)KR_ + 1e-6f);
  for (int i = tid; i < KR_; i += 256)
    ckv[(size_t)row * KR_ + i] = (bf16)(xr[i] * inv * ln_w[i]);
  if (tid < DR_ / 2) {
    int j = tid;
    float p  = (float)pos[row];
    float fr = __powf(10000.0f, -(float)j / 32.0f);
    float c  = __cosf(p * fr), sn = __sinf(p * fr);
    float x0 = xr[KR_ + 2 * j], x1 = xr[KR_ + 2 * j + 1];
    kpe[(size_t)row * DR_ + j]      = x0 * c - x1 * sn;
    kpe[(size_t)row * DR_ + 32 + j] = x0 * sn + x1 * c;
  }
}

// q f32 [S][N*192] -> qf bf16 [N][S][192] with rope on last 64
__global__ void q_process_kernel(const float* __restrict__ q,
                                 const int* __restrict__ pos,
                                 bf16* __restrict__ qfb) {
  const int s = blockIdx.x, n = blockIdx.y, t = threadIdx.x;  // 128 threads
  const float* qp = q + ((size_t)s * N_HEADS + n) * QD_;
  bf16* dst = qfb + ((size_t)n * S_LEN + s) * QD_;
  dst[t] = (bf16)qp[t];  // nope 0..127
  if (t < DR_ / 2) {
    float p  = (float)pos[s];
    float fr = __powf(10000.0f, -(float)t / 32.0f);
    float c  = __cosf(p * fr), sn = __sinf(p * fr);
    float x0 = qp[DN_ + 2 * t], x1 = qp[DN_ + 2 * t + 1];
    dst[DN_ + t]      = (bf16)(x0 * c - x1 * sn);
    dst[DN_ + 32 + t] = (bf16)(x0 * sn + x1 * c);
  }
}

// kv f32 [S][N*256] + kpe f32 [S][64] -> kf bf16 [N][S][192], v bf16 [N][S][128]
__global__ void kv_process_kernel(const float* __restrict__ kv,
                                  const float* __restrict__ kpe,
                                  bf16* __restrict__ kfb,
                                  bf16* __restrict__ vfb) {
  const int s = blockIdx.x, n = blockIdx.y, t = threadIdx.x;  // 128 threads
  const float* kvp = kv + ((size_t)s * N_HEADS + n) * (DN_ + DV_);
  bf16* kd = kfb + ((size_t)n * S_LEN + s) * QD_;
  bf16* vd = vfb + ((size_t)n * S_LEN + s) * DV_;
  kd[t] = (bf16)kvp[t];
  vd[t] = (bf16)kvp[DN_ + t];
  if (t < DR_) kd[DN_ + t] = (bf16)kpe[(size_t)s * DR_ + t];
}

// ---------------------------------------------------------------------------
// Flash attention: one block = 64 query rows of one head; 8 waves.
// K tile staged with async global->LDS; online softmax state in LDS.
// ---------------------------------------------------------------------------
__global__ __launch_bounds__(256) void mla_attn_kernel(const bf16* __restrict__ qf,
                                                       const bf16* __restrict__ kf,
                                                       const bf16* __restrict__ vf,
                                                       bf16* __restrict__ attn_out) {
  __shared__ bf16  Ks[64][QD_];   // 24 KB
  __shared__ bf16  Vt[128][64];   // 16 KB, V transposed [dv][key]
  __shared__ float Ss[64][64];    // 16 KB, scores then probs (f32)
  __shared__ float m_s[64], l_s[64], sc_s[64];

  const int tid  = threadIdx.x;
  const int wid  = tid >> 5;
  const int lane = tid & 31;
  const int qb   = blockIdx.x * 64;
  const int head = blockIdx.y;

  const bf16* Qh = qf + (size_t)head * S_LEN * QD_ + (size_t)qb * QD_;
  const bf16* Kh = kf + (size_t)head * S_LEN * QD_;
  const bf16* Vh = vf + (size_t)head * S_LEN * DV_;

  if (tid < 64) { m_s[tid] = -3.4e38f; l_s[tid] = 0.0f; }

  v8f acc[4];
#pragma unroll
  for (int ni = 0; ni < 4; ni++)
#pragma unroll
    for (int r = 0; r < 8; r++) acc[ni][r] = 0.0f;
  __syncthreads();

  const int tm  = (wid & 3) * 16;    // wave's m-tile rows
  const int qn2 = (wid >> 2) * 32;   // wave's S-col group (2 n-tiles)
  const int on  = (wid >> 2) * 64;   // wave's O-col group (4 n-tiles)

  for (int kb = 0; kb < qb + 64; kb += 64) {
    // async-stage K tile [64][192]; V transposed via VGPRs
    for (int i = tid; i < 64 * 24; i += 256) {
      int r = i / 24, sg = (i % 24) * 8;
      async_load16(Kh + (size_t)(kb + r) * QD_ + sg, &Ks[r][sg]);
    }
    for (int i = tid; i < 64 * 128; i += 256) {
      int k = i >> 7, d = i & 127;
      Vt[d][k] = Vh[(size_t)(kb + k) * DV_ + d];
    }
    wait_async0();
    __syncthreads();

    // S = Q * K^T  (d = 192 -> 6 WMMA k-steps)
    v8f sacc[2];
#pragma unroll
    for (int j = 0; j < 2; j++)
#pragma unroll
      for (int r = 0; r < 8; r++) sacc[j][r] = 0.0f;
    for (int ks = 0; ks < QD_; ks += 32) {
      v16bf aq = frag_b16(Qh, QD_, tm, ks, lane);          // Q direct from global (L2-hot)
      v16bf b0 = frag_b16(&Ks[0][0], QD_, qn2, ks, lane);
      v16bf b1 = frag_b16(&Ks[0][0], QD_, qn2 + 16, ks, lane);
      sacc[0] = WMMA_BF16(aq, b0, sacc[0]);
      sacc[1] = WMMA_BF16(aq, b1, sacc[1]);
    }
#pragma unroll
    for (int j = 0; j < 2; j++)
#pragma unroll
      for (int r = 0; r < 8; r++) {
        int row = tm + r + ((lane >> 4) << 3);
        int col = qn2 + 16 * j + (lane & 15);
        float sv = sacc[j][r] * SCALE_F;
        if (kb + col > qb + row) sv = -3.0e38f;  // causal mask
        Ss[row][col] = sv;
      }
    __syncthreads();

    // online softmax, one thread per row; P written back into Ss as f32
    if (tid < 64) {
      int r = tid;
      float mo = m_s[r];
      float mx = mo;
      for (int c = 0; c < 64; c++) mx = fmaxf(mx, Ss[r][c]);
      float corr = __expf(mo - mx);
      float sum = 0.0f;
      for (int c = 0; c < 64; c++) {
        float p = __expf(Ss[r][c] - mx);
        Ss[r][c] = p;
        sum += p;
      }
      l_s[r] = l_s[r] * corr + sum;
      m_s[r] = mx;
      sc_s[r] = corr;
    }
    __syncthreads();

    // rescale O, then O += P * V
    float scv[8];
#pragma unroll
    for (int r = 0; r < 8; r++) scv[r] = sc_s[tm + r + ((lane >> 4) << 3)];
#pragma unroll
    for (int ni = 0; ni < 4; ni++)
#pragma unroll
      for (int r = 0; r < 8; r++) acc[ni][r] *= scv[r];

    for (int ks2 = 0; ks2 < 64; ks2 += 32) {
      v16bf ap = frag_f32(&Ss[0][0], 64, tm, ks2, lane);
#pragma unroll
      for (int ni = 0; ni < 4; ni++) {
        v16bf bv = frag_b16(&Vt[0][0], 64, on + 16 * ni, ks2, lane);
        acc[ni] = WMMA_BF16(ap, bv, acc[ni]);
      }
    }
    __syncthreads();
  }

#pragma unroll
  for (int ni = 0; ni < 4; ni++)
#pragma unroll
    for (int r = 0; r < 8; r++) {
      int row = tm + r + ((lane >> 4) << 3);
      int col = on + 16 * ni + (lane & 15);
      float o = acc[ni][r] / l_s[row];
      attn_out[(size_t)(qb + row) * (N_HEADS * DV_) + head * DV_ + col] = (bf16)o;
    }
}

// ---------------------------------------------------------------------------
// Host orchestration
// ---------------------------------------------------------------------------
extern "C" void kernel_launch(void* const* d_in, const int* in_sizes, int n_in,
                              void* d_out, int out_size, void* d_ws, size_t ws_size,
                              hipStream_t stream) {
  const float* x       = (const float*)d_in[0];
  const int*   pos     = (const int*)d_in[1];
  const float* q_a_w   = (const float*)d_in[2];
  const float* q_a_ln  = (const float*)d_in[3];
  const float* q_b_w   = (const float*)d_in[4];
  const float* kv_a_w  = (const float*)d_in[5];
  const float* kv_a_ln = (const float*)d_in[6];
  const float* kv_b_w  = (const float*)d_in[7];
  const float* o_w     = (const float*)d_in[8];

  char* base = (char*)d_ws;
  size_t off = 0;
  auto alc = [&](size_t bytes) -> void* {
    void* p = base + off;
    off += (bytes + 255) & ~(size_t)255;
    return p;
  };

  bf16*  xb    = (bf16*) alc((size_t)S_LEN * H_DIM * 2);
  bf16*  wbuf  = (bf16*) alc((size_t)H_DIM * H_DIM * 2);  // reused for each weight
  float* qa_f  = (float*)alc((size_t)S_LEN * QR_ * 4);
  bf16*  qan   = (bf16*) alc((size_t)S_LEN * QR_ * 2);
  float* q_f   = (float*)alc((size_t)S_LEN * N_HEADS * QD_ * 4);
  float* kva_f = (float*)alc((size_t)S_LEN * (KR_ + DR_) * 4);
  bf16*  ckv   = (bf16*) alc((size_t)S_LEN * KR_ * 2);
  float* kpe   = (float*)alc((size_t)S_LEN * DR_ * 4);
  float* kv_f  = (float*)alc((size_t)S_LEN * N_HEADS * (DN_ + DV_) * 4);
  bf16*  qfb   = (bf16*) alc((size_t)N_HEADS * S_LEN * QD_ * 2);
  bf16*  kfb   = (bf16*) alc((size_t)N_HEADS * S_LEN * QD_ * 2);
  bf16*  vfb   = (bf16*) alc((size_t)N_HEADS * S_LEN * DV_ * 2);
  bf16*  attn  = (bf16*) alc((size_t)S_LEN * N_HEADS * DV_ * 2);

  auto cvt = [&](const float* in, bf16* out, size_t n) {
    cvt_f32_bf16_kernel<<<(unsigned)((n + 255) / 256), 256, 0, stream>>>(in, out, (int)n);
  };
  auto gemm = [&](const bf16* A, const bf16* Bm, float* C, int M, int N, int K) {
    dim3 g((unsigned)((N + 127) / 128), (unsigned)((M + 127) / 128));
    gemm_bf16_nt<<<g, 256, 0, stream>>>(A, Bm, C, M, N, K);
  };

  // 1. stage x in bf16
  cvt(x, xb, (size_t)S_LEN * H_DIM);
  // 2. q_a = x @ q_a_w^T ; rmsnorm
  cvt(q_a_w, wbuf, (size_t)QR_ * H_DIM);
  gemm(xb, wbuf, qa_f, S_LEN, QR_, H_DIM);
  rmsnorm_kernel<<<S_LEN, 256, 0, stream>>>(qa_f, q_a_ln, qan, QR_);
  // 3. q = qan @ q_b_w^T
  cvt(q_b_w, wbuf, (size_t)N_HEADS * QD_ * QR_);
  gemm(qan, wbuf, q_f, S_LEN, N_HEADS * QD_, QR_);
  // 4. kv_a = x @ kv_a_w^T ; rmsnorm + k_pe rope
  cvt(kv_a_w, wbuf, (size_t)(KR_ + DR_) * H_DIM);
  gemm(xb, wbuf, kva_f, S_LEN, KR_ + DR_, H_DIM);
  kva_process_kernel<<<S_LEN, 256, 0, stream>>>(kva_f, kv_a_ln, pos, ckv, kpe);
  // 5. kv = ckv @ kv_b_w^T
  cvt(kv_b_w, wbuf, (size_t)N_HEADS * (DN_ + DV_) * KR_);
  gemm(ckv, wbuf, kv_f, S_LEN, N_HEADS * (DN_ + DV_), KR_);
  // 6. repack head-major + rope q_pe
  q_process_kernel<<<dim3(S_LEN, N_HEADS), 128, 0, stream>>>(q_f, pos, qfb);
  kv_process_kernel<<<dim3(S_LEN, N_HEADS), 128, 0, stream>>>(kv_f, kpe, kfb, vfb);
  // 7. flash attention
  mla_attn_kernel<<<dim3(S_LEN / 64, N_HEADS), 256, 0, stream>>>(qfb, kfb, vfb, attn);
  // 8. out = attn @ o_w^T  (f32 straight into d_out)
  cvt(o_w, wbuf, (size_t)H_DIM * N_HEADS * DV_);
  gemm(attn, wbuf, (float*)d_out, S_LEN, H_DIM, N_HEADS * DV_);

  (void)in_sizes; (void)n_in; (void)out_size; (void)ws_size;
}
